// RecoveryODENetwork_14328010899998
// MI455X (gfx1250) — compile-verified, matrix operand order checked by reading, not run on hardware
//
#include <hip/hip_runtime.h>
#include <hip/hip_bf16.h>

// ---- problem dims (from reference) ----
#define B_DIM   512
#define T_DIM   128
#define IN_DIM  128
#define XH_DIM  512
#define G_DIM   256
#define HID_DIM 256
#define OUT_DIM 128
#define BT_DIM  (B_DIM * T_DIM)     // 65536
#define KSUB    4
#define DT_STEP 0.25f

typedef __attribute__((ext_vector_type(16))) __bf16 bf16x16;
typedef __attribute__((ext_vector_type(8)))  float  f32x8;
typedef __attribute__((ext_vector_type(4)))  unsigned int u32x4;
typedef __attribute__((ext_vector_type(4)))  int    i32x4;
typedef __attribute__((ext_vector_type(8)))  int    i32x8;

#if defined(__has_builtin)
#if __has_builtin(__builtin_amdgcn_tensor_load_to_lds)
#define HAVE_TDM 1
#endif
#if __has_builtin(__builtin_amdgcn_global_load_async_to_lds_b128)
#define HAVE_ASYNC_LDS 1
#endif
#endif

__device__ __forceinline__ float sigmoidf(float x) { return 1.0f / (1.0f + __expf(-x)); }

__device__ __forceinline__ f32x8 wmma_bf16(bf16x16 a, bf16x16 b, f32x8 c) {
  return __builtin_amdgcn_wmma_f32_16x16x32_bf16(false, a, false, b, (short)0, c, false, false);
}

// A-matrix fragment (16x32 bf16) from a row-major bf16 buffer.
__device__ __forceinline__ bf16x16 load_frag_a(const __bf16* p, int ld) {
  int lane = threadIdx.x & 31;
  int row = lane & 15, hi = lane >> 4;
  const __bf16* r = p + (size_t)row * ld;
  bf16x16 f;
#pragma unroll
  for (int v = 0; v < 8; ++v) {
    int k = (v >> 2) * 16 + hi * 8 + (v & 3) * 2;
    f[2 * v]     = r[k];
    f[2 * v + 1] = r[k + 1];
  }
  return f;
}

// B-matrix fragment (32x16 bf16) from the pre-swizzled packed weight layout.
__device__ __forceinline__ bf16x16 load_frag_b(const __bf16* packed, int tk, int tn, int NT) {
  int lane = threadIdx.x & 31;
  const __bf16* p = packed + (((size_t)(tk * NT + tn)) * 32 + lane) * 16;
  return *(const bf16x16*)p;
}

// ---------------------------------------------------------------------------
// Weight prep: f32 -> bf16, optional transpose, swizzled into per-lane
// WMMA B-fragment order. trans==0: B[k][n] = src[k*N+n]; trans==1: src[n*K+k].
// ---------------------------------------------------------------------------
__global__ void pack_b_kernel(const float* __restrict__ src, __bf16* __restrict__ dst,
                              int K, int N, int trans) {
  int total = K * N;
  int NT = N >> 4;
  for (int e = blockIdx.x * blockDim.x + threadIdx.x; e < total;
       e += gridDim.x * blockDim.x) {
    int j     = e & 15;
    int per16 = e >> 4;
    int lane  = per16 & 31;
    int tile  = per16 >> 5;
    int tn = tile % NT, tk = tile / NT;
    int v = j >> 1, pair = j & 1, hi = lane >> 4;
    int nn = tn * 16 + (lane & 15);
    int kk = tk * 32 + (v >> 2) * 16 + hi * 8 + (v & 3) * 2 + pair;
    float x = trans ? src[(size_t)nn * K + kk] : src[(size_t)kk * N + nn];
    dst[e] = (__bf16)x;
  }
}

// ---------------------------------------------------------------------------
// Fused x_model: HH = relu(H @ W1 + b1) @ W2 + b2, stored bf16 [BT, G].
// ---------------------------------------------------------------------------
__global__ __launch_bounds__(256) void xmodel_kernel(
    const float* __restrict__ Hg, const float* __restrict__ b1,
    const float* __restrict__ b2, const __bf16* __restrict__ W1p,
    const __bf16* __restrict__ W2p, __bf16* __restrict__ HHg) {
  __shared__ __bf16 mid[32 * XH_DIM];    // 32 KB
  __shared__ __bf16 asta[32 * IN_DIM];   //  8 KB
  int row0 = blockIdx.x * 32;
  int tid = threadIdx.x, lane = tid & 31, w = tid >> 5;

  for (int e = tid; e < 32 * IN_DIM; e += 256) {
    int r = e >> 7, c = e & (IN_DIM - 1);
    asta[e] = (__bf16)Hg[(size_t)(row0 + r) * IN_DIM + c];
  }
  __syncthreads();

  // phase 1: (32x128)(128x512)
  {
    int rb = (w & 1) * 16;
    int cq = (w >> 1) * 128;
    f32x8 acc[8] = {};
    for (int tk = 0; tk < IN_DIM / 32; ++tk) {
      bf16x16 a = load_frag_a(&asta[rb * IN_DIM + tk * 32], IN_DIM);
#pragma unroll
      for (int j = 0; j < 8; ++j) {
        bf16x16 b = load_frag_b(W1p, tk, cq / 16 + j, XH_DIM / 16);
        acc[j] = wmma_bf16(a, b, acc[j]);
      }
    }
#pragma unroll
    for (int j = 0; j < 8; ++j)
#pragma unroll
      for (int i = 0; i < 8; ++i) {
        int rr = rb + i + (lane >> 4) * 8;
        int cc = cq + j * 16 + (lane & 15);
        float v = acc[j][i] + b1[cc];
        mid[rr * XH_DIM + cc] = (__bf16)fmaxf(v, 0.0f);
      }
  }
  __syncthreads();

  // phase 2: (32x512)(512x256)
  {
    int rb = (w & 1) * 16;
    int cq = (w >> 1) * 64;
    f32x8 acc[4] = {};
    for (int tk = 0; tk < XH_DIM / 32; ++tk) {
      bf16x16 a = load_frag_a(&mid[rb * XH_DIM + tk * 32], XH_DIM);
#pragma unroll
      for (int j = 0; j < 4; ++j) {
        bf16x16 b = load_frag_b(W2p, tk, cq / 16 + j, G_DIM / 16);
        acc[j] = wmma_bf16(a, b, acc[j]);
      }
    }
#pragma unroll
    for (int j = 0; j < 4; ++j)
#pragma unroll
      for (int i = 0; i < 8; ++i) {
        int rr = rb + i + (lane >> 4) * 8;
        int cc = cq + j * 16 + (lane & 15);
        HHg[(size_t)(row0 + rr) * G_DIM + cc] = (__bf16)(acc[j][i] + b2[cc]);
      }
  }
}

// ---------------------------------------------------------------------------
// Sequential GRU-ODE recurrence. 32 workgroups x 16 batch rows, 8 waves.
// x_t tiles are DMA'd into LDS by the Tensor Data Mover at the start of each step
// (overlapping the 4 Euler substeps), synchronized with s_wait_tensorcnt.
// ---------------------------------------------------------------------------
__global__ __launch_bounds__(256) void recurrence_kernel(
    const __bf16* __restrict__ HHg, const __bf16* __restrict__ OhrTp,
    const __bf16* __restrict__ OhzTp, const __bf16* __restrict__ OhhTp,
    const __bf16* __restrict__ WihTp, const __bf16* __restrict__ WhhTp,
    const float* __restrict__ bih, const float* __restrict__ bhh,
    __bf16* __restrict__ Houtg) {
  __shared__ float  hbuf[16 * HID_DIM];   // 16 KB  hidden state (f32)
  __shared__ __bf16 hb[16 * HID_DIM];     //  8 KB  bf16(h)
  __shared__ __bf16 sb[16 * HID_DIM];     //  8 KB  r*h substep / r-gate sum
  __shared__ __bf16 zb[16 * HID_DIM];     //  8 KB  z-gate sum
  __shared__ __bf16 xnx[16 * HID_DIM];    //  8 KB  x_t staging, then n-gate x part
  __shared__ __bf16 nhb[16 * HID_DIM];    //  8 KB  n-gate h part
  int tid = threadIdx.x, lane = tid & 31, w = tid >> 5;
  int b0 = blockIdx.x * 16;

  __builtin_prefetch(OhrTp, 0, 3);
  __builtin_prefetch(OhzTp, 0, 3);
  __builtin_prefetch(OhhTp, 0, 3);

  for (int e = tid; e < 16 * HID_DIM; e += 256) hbuf[e] = 0.0f;
  __syncthreads();

  for (int t = 0; t < T_DIM; ++t) {
#if defined(HAVE_TDM)
    // Kick off the TDM 2D tile load of x_t = HH[b0:b0+16, t, :] into LDS now;
    // it overlaps the 4 Euler substeps below (tracked by TENSORcnt).
    if (w == 0) {
      unsigned lds_off = (unsigned)(size_t)(void*)xnx;
      unsigned long long ga =
          (unsigned long long)(size_t)(const void*)(HHg + ((size_t)b0 * T_DIM + t) * G_DIM);
      u32x4 g0;
      g0[0] = 1u;                                   // count=1 (user descriptor)
      g0[1] = lds_off;                              // lds_addr
      g0[2] = (unsigned)(ga & 0xffffffffu);         // global_addr[31:0]
      g0[3] = (unsigned)((ga >> 32) & 0x01ffffffu)  // global_addr[56:32]
              | 0x80000000u;                        // type=2 ("image")
      i32x8 g1;
      g1[0] = 0x00010000;                 // data_size=1 -> 2 bytes/elem
      g1[1] = (int)(G_DIM << 16);         // tensor_dim0[15:0]=256 @ bits[63:48]
      g1[2] = (int)(16 << 16);            // tensor_dim1[15:0]=16  @ bits[95:80]
      g1[3] = (int)(G_DIM << 16);         // tile_dim0=256         @ bits[127:112]
      g1[4] = 16;                         // tile_dim1=16
      g1[5] = T_DIM * G_DIM;              // tensor_dim0_stride[31:0] (row pitch, elems)
      g1[6] = 0;
      g1[7] = 0;
      i32x4 z4 = {};
#if __clang_major__ >= 23
      i32x8 z8 = {};
      __builtin_amdgcn_tensor_load_to_lds(g0, g1, z4, z4, z8, 0);
#else
      __builtin_amdgcn_tensor_load_to_lds(g0, g1, z4, z4, 0);
#endif
    }
#endif

    // ---- KSUB Euler substeps of the autonomous GRU-ODE cell ----
    for (int s = 0; s < KSUB; ++s) {
      for (int e = tid; e < 16 * HID_DIM; e += 256) hb[e] = (__bf16)hbuf[e];
      __syncthreads();
      int c0 = w * 32;

      f32x8 racc[2] = {};
      for (int tk = 0; tk < HID_DIM / 32; ++tk) {
        bf16x16 a = load_frag_a(&hb[tk * 32], HID_DIM);
#pragma unroll
        for (int j = 0; j < 2; ++j) {
          bf16x16 b = load_frag_b(OhrTp, tk, c0 / 16 + j, HID_DIM / 16);
          racc[j] = wmma_bf16(a, b, racc[j]);
        }
      }
#pragma unroll
      for (int j = 0; j < 2; ++j)
#pragma unroll
        for (int i = 0; i < 8; ++i) {
          int rr = i + (lane >> 4) * 8;
          int cc = c0 + j * 16 + (lane & 15);
          float rv = sigmoidf(racc[j][i]);
          sb[rr * HID_DIM + cc] = (__bf16)(rv * hbuf[rr * HID_DIM + cc]);
        }
      __syncthreads();

      f32x8 zacc[2] = {}, uacc[2] = {};
      for (int tk = 0; tk < HID_DIM / 32; ++tk) {
        bf16x16 ah = load_frag_a(&hb[tk * 32], HID_DIM);
        bf16x16 ar = load_frag_a(&sb[tk * 32], HID_DIM);
#pragma unroll
        for (int j = 0; j < 2; ++j) {
          bf16x16 bz = load_frag_b(OhzTp, tk, c0 / 16 + j, HID_DIM / 16);
          zacc[j] = wmma_bf16(ah, bz, zacc[j]);
          bf16x16 bu = load_frag_b(OhhTp, tk, c0 / 16 + j, HID_DIM / 16);
          uacc[j] = wmma_bf16(ar, bu, uacc[j]);
        }
      }
#pragma unroll
      for (int j = 0; j < 2; ++j)
#pragma unroll
        for (int i = 0; i < 8; ++i) {
          int rr = i + (lane >> 4) * 8;
          int cc = c0 + j * 16 + (lane & 15);
          float z = sigmoidf(zacc[j][i]);
          float u = tanhf(uacc[j][i]);
          float ho = hbuf[rr * HID_DIM + cc];
          hbuf[rr * HID_DIM + cc] = ho + DT_STEP * (1.0f - z) * (u - ho);
        }
      __syncthreads();
    }

    // ---- GRU observation step with x_t = HH[:, t, :] ----
    for (int e = tid; e < 16 * HID_DIM; e += 256) hb[e] = (__bf16)hbuf[e];
#if defined(HAVE_TDM)
    if (w == 0) __builtin_amdgcn_s_wait_tensorcnt(0);  // x_t tile landed in LDS
#endif
    __syncthreads();
    {
      int c0 = w * 96;
      f32x8 xacc[6] = {}, hacc[6] = {};
      for (int tk = 0; tk < G_DIM / 32; ++tk) {
        bf16x16 ax;
#if defined(HAVE_TDM)
        ax = load_frag_a(&xnx[tk * 32], HID_DIM);
#else
        {
          int row = lane & 15, hi = lane >> 4;
          const __bf16* r = HHg + ((size_t)(b0 + row) * T_DIM + t) * G_DIM + tk * 32;
#pragma unroll
          for (int v = 0; v < 8; ++v) {
            int k = (v >> 2) * 16 + hi * 8 + (v & 3) * 2;
            ax[2 * v] = r[k];
            ax[2 * v + 1] = r[k + 1];
          }
        }
#endif
        bf16x16 ah = load_frag_a(&hb[tk * 32], HID_DIM);
#pragma unroll
        for (int j = 0; j < 6; ++j) {
          bf16x16 bx = load_frag_b(WihTp, tk, c0 / 16 + j, 768 / 16);
          xacc[j] = wmma_bf16(ax, bx, xacc[j]);
          bf16x16 bh = load_frag_b(WhhTp, tk, c0 / 16 + j, 768 / 16);
          hacc[j] = wmma_bf16(ah, bh, hacc[j]);
        }
      }
      __syncthreads();  // xnx fully consumed; safe to reuse as n-gate x buffer
#pragma unroll
      for (int j = 0; j < 6; ++j)
#pragma unroll
        for (int i = 0; i < 8; ++i) {
          int rr = i + (lane >> 4) * 8;
          int cc = c0 + j * 16 + (lane & 15);
          float xv = xacc[j][i], hv = hacc[j][i];
          if (cc < 256)        sb[rr * HID_DIM + cc] = (__bf16)(xv + hv);
          else if (cc < 512)   zb[rr * HID_DIM + (cc - 256)] = (__bf16)(xv + hv);
          else {
            xnx[rr * HID_DIM + (cc - 512)] = (__bf16)xv;
            nhb[rr * HID_DIM + (cc - 512)] = (__bf16)hv;
          }
        }
    }
    __syncthreads();

    for (int e = tid; e < 16 * HID_DIM; e += 256) {
      int rr = e >> 8, cc = e & (HID_DIM - 1);
      float rsum = (float)sb[e] + bih[cc] + bhh[cc];
      float zsum = (float)zb[e] + bih[256 + cc] + bhh[256 + cc];
      float nx   = (float)xnx[e] + bih[512 + cc];
      float nh   = (float)nhb[e] + bhh[512 + cc];
      float rg = sigmoidf(rsum), zg = sigmoidf(zsum);
      float n  = tanhf(nx + rg * nh);
      float ho = hbuf[e];
      float hn = (1.0f - zg) * n + zg * ho;
      hbuf[e] = hn;
      Houtg[((size_t)(b0 + rr) * T_DIM + t) * HID_DIM + cc] = (__bf16)hn;
    }
    __syncthreads();
  }
}

// ---------------------------------------------------------------------------
// Out = Hout @ Wr + br  (f32 output). Async global->LDS staging of the A tile.
// ---------------------------------------------------------------------------
__global__ __launch_bounds__(128) void out_kernel(
    const __bf16* __restrict__ Houtg, const __bf16* __restrict__ Wrp,
    const float* __restrict__ br, float* __restrict__ Outg) {
  __shared__ __bf16 asta[32 * HID_DIM];  // 16 KB
  int row0 = blockIdx.x * 32;
  int tid = threadIdx.x, lane = tid & 31, w = tid >> 5;
#if defined(HAVE_ASYNC_LDS)
  for (int e = tid; e < (32 * HID_DIM) / 8; e += 128) {
    const __bf16* gp = Houtg + (size_t)row0 * HID_DIM + (size_t)e * 8;
    __bf16* lp = asta + e * 8;
    __builtin_amdgcn_global_load_async_to_lds_b128(
        (__attribute__((address_space(1))) i32x4*)(void*)const_cast<__bf16*>(gp),
        (__attribute__((address_space(3))) i32x4*)(void*)lp, 0, 0);
  }
  asm volatile("s_wait_asynccnt 0" ::: "memory");
#else
  for (int e = tid; e < 32 * HID_DIM; e += 128)
    asta[e] = Houtg[(size_t)row0 * HID_DIM + e];
#endif
  __syncthreads();
  int rb = (w & 1) * 16, cq = (w >> 1) * 64;
  f32x8 acc[4] = {};
  for (int tk = 0; tk < HID_DIM / 32; ++tk) {
    bf16x16 a = load_frag_a(&asta[rb * HID_DIM + tk * 32], HID_DIM);
#pragma unroll
    for (int j = 0; j < 4; ++j) {
      bf16x16 b = load_frag_b(Wrp, tk, cq / 16 + j, OUT_DIM / 16);
      acc[j] = wmma_bf16(a, b, acc[j]);
    }
  }
#pragma unroll
  for (int j = 0; j < 4; ++j)
#pragma unroll
    for (int i = 0; i < 8; ++i) {
      int rr = rb + i + (lane >> 4) * 8;
      int cc = cq + j * 16 + (lane & 15);
      Outg[(size_t)(row0 + rr) * OUT_DIM + cc] = acc[j][i] + br[cc];
    }
}

// ---------------------------------------------------------------------------
// workspace layout (bytes)
// ---------------------------------------------------------------------------
static constexpr size_t OFF_HH   = 0;
static constexpr size_t SZ_HH    = (size_t)BT_DIM * G_DIM * 2;
static constexpr size_t OFF_HOUT = OFF_HH + SZ_HH;
static constexpr size_t SZ_HOUT  = (size_t)BT_DIM * HID_DIM * 2;
static constexpr size_t OFF_W1P  = OFF_HOUT + SZ_HOUT;
static constexpr size_t SZ_W1P   = (size_t)IN_DIM * XH_DIM * 2;
static constexpr size_t OFF_W2P  = OFF_W1P + SZ_W1P;
static constexpr size_t SZ_W2P   = (size_t)XH_DIM * G_DIM * 2;
static constexpr size_t OFF_OHR  = OFF_W2P + SZ_W2P;
static constexpr size_t SZ_OHX   = (size_t)HID_DIM * HID_DIM * 2;
static constexpr size_t OFF_OHZ  = OFF_OHR + SZ_OHX;
static constexpr size_t OFF_OHH  = OFF_OHZ + SZ_OHX;
static constexpr size_t OFF_WIH  = OFF_OHH + SZ_OHX;
static constexpr size_t SZ_WXT   = (size_t)G_DIM * 3 * HID_DIM * 2;
static constexpr size_t OFF_WHH  = OFF_WIH + SZ_WXT;
static constexpr size_t OFF_WRP  = OFF_WHH + SZ_WXT;
static constexpr size_t SZ_WRP   = (size_t)HID_DIM * OUT_DIM * 2;

extern "C" void kernel_launch(void* const* d_in, const int* in_sizes, int n_in,
                              void* d_out, int out_size, void* d_ws, size_t ws_size,
                              hipStream_t stream) {
  const float* Hg  = (const float*)d_in[0];
  const float* W1  = (const float*)d_in[2];
  const float* b1  = (const float*)d_in[3];
  const float* W2  = (const float*)d_in[4];
  const float* b2  = (const float*)d_in[5];
  const float* Ohr = (const float*)d_in[6];
  const float* Ohz = (const float*)d_in[7];
  const float* Ohh = (const float*)d_in[8];
  const float* Wih = (const float*)d_in[9];
  const float* Whh = (const float*)d_in[10];
  const float* bih = (const float*)d_in[11];
  const float* bhh = (const float*)d_in[12];
  const float* Wr  = (const float*)d_in[13];
  const float* br  = (const float*)d_in[14];
  float* Outg = (float*)d_out;

  char* ws = (char*)d_ws;
  __bf16* HHg   = (__bf16*)(ws + OFF_HH);
  __bf16* Houtg = (__bf16*)(ws + OFF_HOUT);
  __bf16* W1p   = (__bf16*)(ws + OFF_W1P);
  __bf16* W2p   = (__bf16*)(ws + OFF_W2P);
  __bf16* OhrTp = (__bf16*)(ws + OFF_OHR);
  __bf16* OhzTp = (__bf16*)(ws + OFF_OHZ);
  __bf16* OhhTp = (__bf16*)(ws + OFF_OHH);
  __bf16* WihTp = (__bf16*)(ws + OFF_WIH);
  __bf16* WhhTp = (__bf16*)(ws + OFF_WHH);
  __bf16* Wrp   = (__bf16*)(ws + OFF_WRP);

  auto pgrid = [](int total) { int g = (total + 255) / 256; return g > 2048 ? 2048 : g; };

  pack_b_kernel<<<pgrid(IN_DIM * XH_DIM), 256, 0, stream>>>(W1, W1p, IN_DIM, XH_DIM, 0);
  pack_b_kernel<<<pgrid(XH_DIM * G_DIM), 256, 0, stream>>>(W2, W2p, XH_DIM, G_DIM, 0);
  pack_b_kernel<<<pgrid(HID_DIM * HID_DIM), 256, 0, stream>>>(Ohr, OhrTp, HID_DIM, HID_DIM, 1);
  pack_b_kernel<<<pgrid(HID_DIM * HID_DIM), 256, 0, stream>>>(Ohz, OhzTp, HID_DIM, HID_DIM, 1);
  pack_b_kernel<<<pgrid(HID_DIM * HID_DIM), 256, 0, stream>>>(Ohh, OhhTp, HID_DIM, HID_DIM, 1);
  pack_b_kernel<<<pgrid(G_DIM * 3 * HID_DIM), 256, 0, stream>>>(Wih, WihTp, G_DIM, 3 * HID_DIM, 1);
  pack_b_kernel<<<pgrid(HID_DIM * 3 * HID_DIM), 256, 0, stream>>>(Whh, WhhTp, HID_DIM, 3 * HID_DIM, 1);
  pack_b_kernel<<<pgrid(HID_DIM * OUT_DIM), 256, 0, stream>>>(Wr, Wrp, HID_DIM, OUT_DIM, 0);

  xmodel_kernel<<<BT_DIM / 32, 256, 0, stream>>>(Hg, b1, b2, W1p, W2p, HHg);

  recurrence_kernel<<<B_DIM / 16, 256, 0, stream>>>(HHg, OhrTp, OhzTp, OhhTp,
                                                    WihTp, WhhTp, bih, bhh, Houtg);

  out_kernel<<<BT_DIM / 32, 128, 0, stream>>>(Houtg, Wrp, br, Outg);
}
